// DeepRNNScratch_4157528343263
// MI455X (gfx1250) — compile-verified, matrix operand order checked by reading, not run on hardware
//
#include <hip/hip_runtime.h>
#include <hip/hip_bf16.h>

// ---------------------------------------------------------------------------
// Deep RNN (L=4, T=256, B=128, H=1024) on MI455X / gfx1250.
//   * bf16 WMMA (v_wmma_f32_16x16x32_bf16) for pre-projection + recurrence
//   * activations kept as bf16 copies so GEMM staging is a pure byte copy
//   * global_load_async_to_lds_b128/b64 (ASYNCcnt) with double-buffered LDS
//   * f32 accumulate, tanh epilogue writes f32 result + bf16 activation
// ---------------------------------------------------------------------------

typedef __attribute__((ext_vector_type(16))) __bf16 v16bf;
typedef __attribute__((ext_vector_type(8)))  float  v8f;
typedef int v4i_t __attribute__((ext_vector_type(4)));
typedef int v2i_t __attribute__((ext_vector_type(2)));

// clang prints addrspace(1) as __device__, addrspace(3) as __shared__ in HIP mode
typedef __attribute__((address_space(1))) v4i_t gm_v4i;
typedef __attribute__((address_space(3))) v4i_t lds_v4i;
typedef __attribute__((address_space(1))) v2i_t gm_v2i;
typedef __attribute__((address_space(3))) v2i_t lds_v2i;

#define T_STEPS 256
#define BATCH   128
#define HID     1024
#define LAYERS  4

#define BM 128        // block tile rows
#define KK 32         // K per WMMA step
#define ASTR 40       // LDS row stride (ushorts) = KK + 8 pad, keeps 16B align
#define BSTR 40

#if defined(__HIP_DEVICE_COMPILE__) && \
    __has_builtin(__builtin_amdgcn_global_load_async_to_lds_b128) && \
    __has_builtin(__builtin_amdgcn_global_load_async_to_lds_b64)
#define HAVE_ASYNC_LDS 1
#else
#define HAVE_ASYNC_LDS 0
#endif

__device__ __forceinline__ unsigned short f2bf(float f) {
  unsigned u = __float_as_uint(f);
  u += 0x7FFFu + ((u >> 16) & 1u);   // round-to-nearest-even
  return (unsigned short)(u >> 16);
}

__device__ __forceinline__ void cp_b128(unsigned short* lds, const unsigned short* g) {
#if HAVE_ASYNC_LDS
  __builtin_amdgcn_global_load_async_to_lds_b128((gm_v4i*)(void*)g, (lds_v4i*)lds, 0, 0);
#else
  *(uint4*)lds = *(const uint4*)g;
#endif
}

__device__ __forceinline__ void cp_b64(unsigned short* lds, const unsigned short* g) {
#if HAVE_ASYNC_LDS
  __builtin_amdgcn_global_load_async_to_lds_b64((gm_v2i*)(void*)g, (lds_v2i*)lds, 0, 0);
#else
  *(uint2*)lds = *(const uint2*)g;
#endif
}

template <int N>
__device__ __forceinline__ void wait_async() {
#if HAVE_ASYNC_LDS
#if __has_builtin(__builtin_amdgcn_s_wait_asynccnt)
  __builtin_amdgcn_s_wait_asynccnt(N);
#else
  asm volatile("s_wait_asynccnt %0" :: "i"(N) : "memory");
#endif
#endif
}

union FragCast { v16bf v; uint4 q[2]; };

// Stage one K-tile (A: BM x KK bf16, B^T: BN x KK bf16) into LDS.
// Uniform 3 async ops per thread: 2x b128 for A, 1x (b128|b64) for B.
template <int NT>
__device__ __forceinline__ void stage_tiles(const unsigned short* __restrict__ A,
                                            const unsigned short* __restrict__ BT,
                                            unsigned short* As, unsigned short* Bs,
                                            int tid, int m0, int n0, int k0) {
#pragma unroll
  for (int r = 0; r < 2; ++r) {
    int idx = tid + 256 * r;            // 512 chunks of 8 ushorts (16B)
    int row = idx >> 2;                 // 4 chunks per 32-wide row
    int c   = (idx & 3) * 8;
    cp_b128(&As[row * ASTR + c], A + (size_t)(m0 + row) * HID + k0 + c);
  }
  if (NT == 2) {                        // BN=64: 256 x 16B chunks
    int row = tid >> 2;
    int c   = (tid & 3) * 8;
    cp_b128(&Bs[row * BSTR + c], BT + (size_t)(n0 + row) * HID + k0 + c);
  } else {                              // BN=32: 256 x 8B chunks
    int row = tid >> 3;
    int c   = (tid & 7) * 4;
    cp_b64(&Bs[row * BSTR + c], BT + (size_t)(n0 + row) * HID + k0 + c);
  }
}

// MODE 0: C = A*B + bias[n]                           (pre-projection)
// MODE 1: v = tanh(A*B + C); C = v; Cbf = bf16(v)     (recurrent step; A may be null => h0=0)
// NT: 16-col tiles per wave (block tile = 128 x 32*NT), 8 wave32 per block.
template <int MODE, int NT>
__global__ __launch_bounds__(256)
void gemm_rnn(const unsigned short* __restrict__ A,    // bf16 [M][H]
              const unsigned short* __restrict__ BT,   // bf16 B^T [N][H]
              const float* __restrict__ bias,
              float* __restrict__ C,
              unsigned short* __restrict__ Cbf) {
  __shared__ unsigned short As[2][BM * ASTR];
  __shared__ unsigned short Bs[2][64 * BSTR];

  const int tid  = threadIdx.x;
  const int lane = tid & 31;           // wave32
  const int wave = tid >> 5;           // 0..7
  const int wrow = wave & 3;           // 4 wave-rows x 32 rows
  const int wcol = wave >> 2;          // 2 wave-cols x NT*16 cols
  constexpr int BN = NT * 32;
  const int m0 = blockIdx.y * BM;
  const int n0 = blockIdx.x * BN;

  v8f acc[2][NT] = {};

  if (A != nullptr) {
    const int fr = lane & 15;
    const int kb = (lane >> 4) * 8;    // per-lane K sub-chunk base (0 or 8)

    stage_tiles<NT>(A, BT, As[0], Bs[0], tid, m0, n0, 0);
    stage_tiles<NT>(A, BT, As[1], Bs[1], tid, m0, n0, KK);

    int cur = 0;
    for (int k0 = 0; k0 < HID; k0 += KK) {
      if (k0 + KK < HID) wait_async<3>();   // one stage (3 ops/thread) may stay in flight
      else               wait_async<0>();
      __syncthreads();

      const unsigned short* Ac = As[cur];
      const unsigned short* Bc = Bs[cur];
      FragCast a0, a1, b[NT];
      {
        const unsigned short* p = &Ac[(wrow * 32 + fr) * ASTR + kb];
        a0.q[0] = *(const uint4*)p;  a0.q[1] = *(const uint4*)(p + 16);
        p = &Ac[(wrow * 32 + 16 + fr) * ASTR + kb];
        a1.q[0] = *(const uint4*)p;  a1.q[1] = *(const uint4*)(p + 16);
      }
#pragma unroll
      for (int j = 0; j < NT; ++j) {
        const unsigned short* p = &Bc[((wcol * NT + j) * 16 + fr) * BSTR + kb];
        b[j].q[0] = *(const uint4*)p;  b[j].q[1] = *(const uint4*)(p + 16);
      }
#pragma unroll
      for (int j = 0; j < NT; ++j) {
        acc[0][j] = __builtin_amdgcn_wmma_f32_16x16x32_bf16(false, a0.v, false, b[j].v, (short)0, acc[0][j], false, false);
        acc[1][j] = __builtin_amdgcn_wmma_f32_16x16x32_bf16(false, a1.v, false, b[j].v, (short)0, acc[1][j], false, false);
      }
      __syncthreads();
      if (k0 + 2 * KK < HID)
        stage_tiles<NT>(A, BT, As[cur], Bs[cur], tid, m0, n0, k0 + 2 * KK);
      cur ^= 1;
    }
  }

  // Epilogue. C/D layout: VGPR v holds M = v (lanes 0-15) / v+8 (lanes 16-31).
  const int colHalf = lane & 15;
  const int rowHalf = (lane >> 4) << 3;
#pragma unroll
  for (int i = 0; i < 2; ++i) {
#pragma unroll
    for (int j = 0; j < NT; ++j) {
      const int col   = n0 + wcol * (NT * 16) + j * 16 + colHalf;
      const int rbase = m0 + wrow * 32 + i * 16 + rowHalf;
#pragma unroll
      for (int v = 0; v < 8; ++v) {
        const size_t off = (size_t)(rbase + v) * HID + col;
        if (MODE == 0) {
          C[off] = acc[i][j][v] + bias[col];
        } else {
          float val = tanhf(acc[i][j][v] + C[off]);
          C[off]   = val;
          Cbf[off] = f2bf(val);
        }
      }
    }
  }
}

// Convert + transpose both weight tensors: out[l][n][k] = bf16(W[l][k][n]).
__global__ void convert_weights(const float* __restrict__ Wxh,
                                const float* __restrict__ Whh,
                                unsigned short* __restrict__ WxhT,
                                unsigned short* __restrict__ WhhT) {
  size_t idx = (size_t)blockIdx.x * blockDim.x + threadIdx.x;
  const size_t total = (size_t)LAYERS * HID * HID;
  if (idx >= total) return;
  size_t l = idx / ((size_t)HID * HID);
  size_t r = idx % ((size_t)HID * HID);
  size_t n = r / HID, k = r % HID;
  size_t src = l * (size_t)HID * HID + k * HID + n;
  WxhT[idx] = f2bf(Wxh[src]);
  WhhT[idx] = f2bf(Whh[src]);
}

__global__ void convert_act(const float* __restrict__ src, unsigned short* __restrict__ dst) {
  size_t i = (size_t)blockIdx.x * blockDim.x + threadIdx.x;   // T*B*H elements
  dst[i] = f2bf(src[i]);
}

__global__ void copy_hlast(const float* __restrict__ src, float* __restrict__ dst) {
  int i = blockIdx.x * blockDim.x + threadIdx.x;              // B*H elements
  dst[i] = src[i];
}

extern "C" void kernel_launch(void* const* d_in, const int* in_sizes, int n_in,
                              void* d_out, int out_size, void* d_ws, size_t ws_size,
                              hipStream_t stream) {
  const float* x   = (const float*)d_in[0];   // [T,B,H]
  const float* Wxh = (const float*)d_in[1];   // [L,H,H]
  const float* Whh = (const float*)d_in[2];   // [L,H,H]
  const float* bh  = (const float*)d_in[3];   // [L,1,H]
  float* out = (float*)d_out;

  const size_t HH   = (size_t)HID * HID;
  const size_t WN   = (size_t)LAYERS * HH;
  const size_t NACT = (size_t)T_STEPS * BATCH * HID;
  const size_t BH   = (size_t)BATCH * HID;

  // ws layout: WxhT bf16 (8MB) | WhhT bf16 (8MB) | Xbf0 (64MB) | Xbf1 (64MB) | xw f32 (134MB)
  unsigned short* WxhT = (unsigned short*)d_ws;
  unsigned short* WhhT = WxhT + WN;
  unsigned short* Xbf0 = WhhT + WN;
  unsigned short* Xbf1 = Xbf0 + NACT;
  float* xwbuf = (float*)(Xbf1 + NACT);

  convert_weights<<<(unsigned)((WN + 255) / 256), 256, 0, stream>>>(Wxh, Whh, WxhT, WhhT);
  convert_act<<<(unsigned)(NACT / 256), 256, 0, stream>>>(x, Xbf0);

  float* outputs = out;                 // first T*B*H floats
  float* Hs      = out + NACT;          // then L*B*H floats

  for (int l = 0; l < LAYERS; ++l) {
    const unsigned short* Ain = (l & 1) ? Xbf1 : Xbf0;   // bf16 layer input sequence
    unsigned short*       Hbf = (l & 1) ? Xbf0 : Xbf1;   // bf16 hidden outputs (next input)
    float* xw = (l < 3) ? xwbuf : outputs;               // f32 xw / h buffer; final -> d_out

    // xw = X @ W_xh + b   (M = T*B = 32768), block tile 128x64
    dim3 g1(HID / 64, (T_STEPS * BATCH) / BM);
    gemm_rnn<0, 2><<<g1, 256, 0, stream>>>(Ain, WxhT + (size_t)l * HH,
                                           bh + (size_t)l * HID, xw, nullptr);

    // recurrence: xw[t] <- tanh(xw[t] + h_{t-1} @ W_hh); also emit bf16 h_t
    const unsigned short* WT = WhhT + (size_t)l * HH;
    dim3 g2(HID / 32, BATCH / BM);   // (32, 1) blocks per step
    for (int t = 0; t < T_STEPS; ++t) {
      const unsigned short* hprev = t ? (Hbf + (size_t)(t - 1) * BH) : nullptr;
      gemm_rnn<1, 1><<<g2, 256, 0, stream>>>(hprev, WT, nullptr,
                                             xw + (size_t)t * BH, Hbf + (size_t)t * BH);
    }

    copy_hlast<<<(unsigned)(BH / 256), 256, 0, stream>>>(
        xw + (size_t)(T_STEPS - 1) * BH, Hs + (size_t)l * BH);
  }
  (void)in_sizes; (void)n_in; (void)out_size; (void)ws_size;
}
